// GroupedQueryAttention_70291434766676
// MI455X (gfx1250) — compile-verified, matrix operand order checked by reading, not run on hardware
//
#include <hip/hip_runtime.h>
#include <hip/hip_bf16.h>

// ---------------------------------------------------------------------------
// GQA prefill for MI455X (gfx1250), bf16 WMMA pipeline, b128 fragment loads,
// TDM (tensor_load_to_lds) staging of K/V tiles in the attention kernel.
// B=2, T=2048, C=2048, H=16, KV=4, D=128, G=4.
// ---------------------------------------------------------------------------

typedef float    v8f   __attribute__((ext_vector_type(8)));
typedef __bf16   bf16_t;
typedef bf16_t   v16bf __attribute__((ext_vector_type(16)));
typedef unsigned u32x4 __attribute__((ext_vector_type(4)));
typedef unsigned v4u   __attribute__((ext_vector_type(4)));
typedef int      v8i   __attribute__((ext_vector_type(8)));
typedef int      v4i   __attribute__((ext_vector_type(4)));

#if defined(__AMDGCN__) && __has_builtin(__builtin_amdgcn_tensor_load_to_lds) \
    && __has_builtin(__builtin_amdgcn_s_wait_tensorcnt)
#define HAVE_TDM 1
#else
#define HAVE_TDM 0
#endif

union Frag {
  unsigned u[8];
  u32x4    q[2];
  v16bf    v;
};

__device__ __forceinline__ unsigned short f2bf(float f) {
  unsigned u = __float_as_uint(f);
  u += 0x7FFFu + ((u >> 16) & 1u);  // round-to-nearest-even
  return (unsigned short)(u >> 16);
}

// Load one 16-bit fragment whose K-dim is contiguous at `row`.
// (dwords 0-3 = K[8*khalf .. +7], dwords 4-7 = K[16+8*khalf .. +7])
template <typename P>
__device__ __forceinline__ void load_frag(Frag& f, P row, int khalf) {
  f.q[0] = *(const u32x4*)(row + 8 * khalf);
  f.q[1] = *(const u32x4*)(row + 16 + 8 * khalf);
}

#if HAVE_TDM
// Issue a TDM 2D tile load (bf16 elements) into LDS. All args wave-uniform.
// tensor_d0/tile_d0: contiguous dim (elements); stride0: line stride (elems).
__device__ __forceinline__ void tdm_load_2d(unsigned lds_off, const void* gptr,
                                            unsigned tensor_d0,
                                            unsigned tensor_d1,
                                            unsigned tile_d0, unsigned tile_d1,
                                            unsigned long long stride0) {
  unsigned long long ga = (unsigned long long)(uintptr_t)gptr;
  v4u g0;
  g0[0] = 1u;                                             // count = 1
  g0[1] = lds_off;                                        // lds_addr (bytes)
  g0[2] = (unsigned)(ga & 0xffffffffu);                   // global_addr lo
  g0[3] = (unsigned)((ga >> 32) & 0x1ffffffu) | (2u << 30);  // addr hi | type=2
  v8i g1;
  g1[0] = 0x10000;  // data_size = 1 (2 bytes), no multicast/pad/iterate
  g1[1] = (int)((tensor_d0 & 0xffffu) << 16);
  g1[2] = (int)(((tensor_d0 >> 16) & 0xffffu) | ((tensor_d1 & 0xffffu) << 16));
  g1[3] = (int)(((tensor_d1 >> 16) & 0xffffu) | ((tile_d0 & 0xffffu) << 16));
  g1[4] = (int)(tile_d1 & 0xffffu);                       // tile_dim2 = 0
  g1[5] = (int)(stride0 & 0xffffffffu);
  g1[6] = (int)((stride0 >> 32) & 0xffffu);               // dim1_stride = 0
  g1[7] = 0;
  v4i z4 = {0, 0, 0, 0};
#if __clang_major__ >= 23
  v8i z8 = {0, 0, 0, 0, 0, 0, 0, 0};
  __builtin_amdgcn_tensor_load_to_lds(g0, g1, z4, z4, z8, 0);
#else
  __builtin_amdgcn_tensor_load_to_lds(g0, g1, z4, z4, 0);
#endif
}
#endif

// ---------------------------------------------------------------------------
// f32 -> bf16 cast (row-major copy)
// ---------------------------------------------------------------------------
__global__ void cast_bf16_kernel(const float* __restrict__ src,
                                 unsigned short* __restrict__ dst, int n) {
  int i = blockIdx.x * 256 + threadIdx.x;
  if (i < n) dst[i] = f2bf(src[i]);
}

// f32 [K,N] -> bf16 [N,K] transpose-cast (weights: makes K contiguous for B)
__global__ void cast_transpose_bf16_kernel(const float* __restrict__ src,
                                           unsigned short* __restrict__ dst,
                                           int K, int N) {
  size_t i = (size_t)blockIdx.x * 256 + threadIdx.x;
  if (i >= (size_t)K * N) return;
  int n = (int)(i % N);
  int k = (int)(i / N);
  dst[(size_t)n * K + k] = f2bf(src[i]);
}

// ---------------------------------------------------------------------------
// bf16 GEMM: C[M,N] = A[M,K] @ Bt[N,K]^T, f32 accumulate.
// block = 128 threads (4 waves in 2x2), wave tile 32x64, block tile 64x128.
// ---------------------------------------------------------------------------
__global__ __launch_bounds__(128) void gemm_bf16_kernel(
    const unsigned short* __restrict__ A, const unsigned short* __restrict__ Bt,
    float* __restrict__ C, int M, int N, int K) {
  const int lane  = threadIdx.x & 31;
  const int wave  = threadIdx.x >> 5;
  const int mlo   = lane & 15;
  const int khalf = lane >> 4;
  const int mBase = blockIdx.y * 64 + (wave >> 1) * 32;
  const int nBase = blockIdx.x * 128 + (wave & 1) * 64;

  v8f acc[2][4] = {};

  for (int k0 = 0; k0 < K; k0 += 32) {
    Frag a[2], b[4];
#pragma unroll
    for (int mt = 0; mt < 2; ++mt) {
      const unsigned short* arow =
          A + (size_t)(mBase + mt * 16 + mlo) * K + k0;
      __builtin_prefetch(arow + 128, 0, 1);
      load_frag(a[mt], arow, khalf);
    }
#pragma unroll
    for (int nt = 0; nt < 4; ++nt) {
      const unsigned short* brow =
          Bt + (size_t)(nBase + nt * 16 + mlo) * K + k0;
      load_frag(b[nt], brow, khalf);
    }
#pragma unroll
    for (int mt = 0; mt < 2; ++mt)
#pragma unroll
      for (int nt = 0; nt < 4; ++nt)
        acc[mt][nt] = __builtin_amdgcn_wmma_f32_16x16x32_bf16(
            false, a[mt].v, false, b[nt].v, (short)0, acc[mt][nt], false,
            false);
  }

#pragma unroll
  for (int mt = 0; mt < 2; ++mt)
#pragma unroll
    for (int nt = 0; nt < 4; ++nt)
#pragma unroll
      for (int r = 0; r < 8; ++r) {
        const int m = mBase + mt * 16 + r + 8 * khalf;
        const int n = nBase + nt * 16 + mlo;
        C[(size_t)m * N + n] = acc[mt][nt][r];
      }
}

// ---------------------------------------------------------------------------
// RoPE + transpose [B,T,NH,D] f32 -> [B,NH,T,D] bf16  (Q and K)
// ---------------------------------------------------------------------------
__global__ void rope_transpose_kernel(const float* __restrict__ src,
                                      unsigned short* __restrict__ dst, int Bsz,
                                      int T, int NH, int D) {
  const size_t total = (size_t)Bsz * T * NH * D;
  const size_t i = (size_t)blockIdx.x * 256 + threadIdx.x;
  if (i >= total) return;
  const int d = (int)(i % D);
  const int h = (int)((i / D) % NH);
  const int t = (int)((i / ((size_t)D * NH)) % T);
  const int b = (int)(i / ((size_t)D * NH * T));

  const float* row = src + ((size_t)(b * T + t) * NH + h) * D;
  const float x = row[d];
  const int j = d & 63;  // d % (D/2)
  const float inv = __powf(10000.0f, -(float)j * (1.0f / 64.0f));
  float s, c;
  __sincosf((float)t * inv, &s, &c);
  const float partner = (d < 64) ? -row[d + 64] : row[d - 64];
  dst[((size_t)(b * NH + h) * T + t) * D + d] = f2bf(x * c + partner * s);
}

// V: [B,T,KV,D] f32 -> [B,KV,D,T] bf16 (key index contiguous => b128 B-frags)
__global__ void v_transpose_kernel(const float* __restrict__ src,
                                   unsigned short* __restrict__ dst, int Bsz,
                                   int T, int KV, int D) {
  const size_t total = (size_t)Bsz * T * KV * D;
  const size_t i = (size_t)blockIdx.x * 256 + threadIdx.x;
  if (i >= total) return;
  const int d  = (int)(i % D);
  const int kv = (int)((i / D) % KV);
  const int t  = (int)((i / ((size_t)D * KV)) % T);
  const int b  = (int)(i / ((size_t)D * KV * T));
  dst[(((size_t)b * KV + kv) * D + d) * T + t] = f2bf(src[i]);
}

// ---------------------------------------------------------------------------
// Flash attention: one wave per (b, h, 16-row q block).
// Q: [B,H,T,D] bf16   K: [B,KV,T,D] bf16   Vt: [B,KV,D,T] bf16
// ctx: [B,T,H*D] bf16
// K/V tiles staged into LDS by the Tensor Data Mover when available.
// ---------------------------------------------------------------------------
__global__ __launch_bounds__(32) void attn_kernel(
    const unsigned short* __restrict__ Q, const unsigned short* __restrict__ Kb,
    const unsigned short* __restrict__ Vt, unsigned short* __restrict__ ctx,
    int T) {
  __shared__ __align__(16) unsigned short Pl[16 * 32];
#if HAVE_TDM
  __shared__ __align__(16) unsigned short Klds[32 * 128];  // [key][d]
  __shared__ __align__(16) unsigned short Vlds[128 * 32];  // [d][key]
#endif

  const int lane  = threadIdx.x & 31;
  const int mlo   = lane & 15;
  const int khalf = lane >> 4;

  const int nqb = T / 16;
  const int qb  = blockIdx.x % nqb;
  const int h   = (blockIdx.x / nqb) % 16;
  const int b   = blockIdx.x / (nqb * 16);
  const int q0  = qb * 16;
  const int kvh = h >> 2;  // G = 4

  const float scale = 0.08838834764831845f;  // 128^-0.5

  const unsigned short* Qp = Q + (((size_t)b * 16 + h) * T + q0) * 128;
  const unsigned short* Kp = Kb + (((size_t)b * 4 + kvh) * T) * 128;
  const unsigned short* Vp = Vt + (((size_t)b * 4 + kvh) * 128) * T;

  // Q fragments (4 chunks of head-dim 32) resident in registers.
  Frag qf[4];
#pragma unroll
  for (int dc = 0; dc < 4; ++dc)
    load_frag(qf[dc], Qp + (size_t)mlo * 128 + dc * 32, khalf);

  v8f o[8] = {};
  float mrow[8], lrow[8];
#pragma unroll
  for (int r = 0; r < 8; ++r) {
    mrow[r] = -1e30f;
    lrow[r] = 0.0f;
  }

  const int kend = q0 + 16;
  for (int k0 = 0; k0 < kend; k0 += 32) {
#if HAVE_TDM
    // ---- TDM: DMA K tile (32 keys x 128) and V tile (128 d x 32 keys) ----
    tdm_load_2d((unsigned)(uintptr_t)Klds, Kp + (size_t)k0 * 128,
                /*tensor_d0=*/128, /*tensor_d1=*/(unsigned)(T - k0),
                /*tile_d0=*/128, /*tile_d1=*/32, /*stride0=*/128ull);
    tdm_load_2d((unsigned)(uintptr_t)Vlds, Vp + k0,
                /*tensor_d0=*/(unsigned)T, /*tensor_d1=*/128,
                /*tile_d0=*/32, /*tile_d1=*/128, /*stride0=*/(unsigned long long)T);
    __builtin_amdgcn_s_wait_tensorcnt(0);
    __syncthreads();
#endif

    // ---- S = (Q @ K^T) * scale, two 16x16 tiles ----
    v8f s[2] = {};
#pragma unroll
    for (int nt = 0; nt < 2; ++nt) {
#pragma unroll
      for (int dc = 0; dc < 4; ++dc) {
        Frag kf;
#if HAVE_TDM
        load_frag(kf, (const unsigned short*)Klds +
                          (size_t)(nt * 16 + mlo) * 128 + dc * 32,
                  khalf);
#else
        load_frag(kf, Kp + (size_t)(k0 + nt * 16 + mlo) * 128 + dc * 32,
                  khalf);
#endif
        s[nt] = __builtin_amdgcn_wmma_f32_16x16x32_bf16(
            false, qf[dc].v, false, kf.v, (short)0, s[nt], false, false);
      }
    }

    // ---- scale + causal mask ----
#pragma unroll
    for (int nt = 0; nt < 2; ++nt)
#pragma unroll
      for (int r = 0; r < 8; ++r) {
        const int qrow = q0 + r + 8 * khalf;
        const int kcol = k0 + nt * 16 + mlo;
        float val = s[nt][r] * scale;
        if (kcol > qrow) val = -1e30f;
        s[nt][r] = val;
      }

    // ---- online softmax (row spans 16 lanes within each half-wave) ----
    float alpha[8];
#pragma unroll
    for (int r = 0; r < 8; ++r) {
      float mx = fmaxf(s[0][r], s[1][r]);
#pragma unroll
      for (int off = 1; off < 16; off <<= 1)
        mx = fmaxf(mx, __shfl_xor(mx, off, 32));
      const float mnew = fmaxf(mrow[r], mx);
      alpha[r] = __expf(mrow[r] - mnew);
      mrow[r] = mnew;
    }
#pragma unroll
    for (int dt = 0; dt < 8; ++dt)
#pragma unroll
      for (int r = 0; r < 8; ++r) o[dt][r] *= alpha[r];

#pragma unroll
    for (int nt = 0; nt < 2; ++nt)
#pragma unroll
      for (int r = 0; r < 8; ++r) s[nt][r] = __expf(s[nt][r] - mrow[r]);

#pragma unroll
    for (int r = 0; r < 8; ++r) {
      float rs = s[0][r] + s[1][r];
#pragma unroll
      for (int off = 1; off < 16; off <<= 1) rs += __shfl_xor(rs, off, 32);
      lrow[r] = lrow[r] * alpha[r] + rs;
    }

    // ---- P (C-layout f32) -> LDS bf16 16x32 -> A-layout fragment ----
#pragma unroll
    for (int nt = 0; nt < 2; ++nt)
#pragma unroll
      for (int r = 0; r < 8; ++r)
        Pl[(r + 8 * khalf) * 32 + nt * 16 + mlo] = f2bf(s[nt][r]);
    __syncthreads();

    Frag pf;
    pf.q[0] = *(const u32x4*)&Pl[mlo * 32 + 8 * khalf];
    pf.q[1] = *(const u32x4*)&Pl[mlo * 32 + 16 + 8 * khalf];

    // ---- O += P @ V : V rows contiguous in key index ----
#pragma unroll
    for (int dt = 0; dt < 8; ++dt) {
      Frag vf;
#if HAVE_TDM
      load_frag(vf, (const unsigned short*)Vlds + (size_t)(dt * 16 + mlo) * 32,
                khalf);
#else
      load_frag(vf, Vp + (size_t)(dt * 16 + mlo) * T + k0, khalf);
#endif
      o[dt] = __builtin_amdgcn_wmma_f32_16x16x32_bf16(
          false, pf.v, false, vf.v, (short)0, o[dt], false, false);
    }
    __syncthreads();
  }

  // ---- epilogue: normalize, write ctx[b, t, h*128 + d] bf16 ----
#pragma unroll
  for (int dt = 0; dt < 8; ++dt)
#pragma unroll
    for (int r = 0; r < 8; ++r) {
      const int t = q0 + r + 8 * khalf;
      const float val = o[dt][r] / lrow[r];
      ctx[((size_t)b * T + t) * 2048 + h * 128 + dt * 16 + mlo] = f2bf(val);
    }
}

// ---------------------------------------------------------------------------
// Host launcher
// ---------------------------------------------------------------------------
extern "C" void kernel_launch(void* const* d_in, const int* in_sizes, int n_in,
                              void* d_out, int out_size, void* d_ws,
                              size_t ws_size, hipStream_t stream) {
  (void)in_sizes; (void)n_in; (void)out_size; (void)ws_size;

  constexpr int B = 2, T = 2048, C = 2048, H = 16, KV = 4, D = 128;
  constexpr int M = B * T;     // 4096
  constexpr int Nq = H * D;    // 2048
  constexpr int Nkv = KV * D;  // 512

  const float* x  = (const float*)d_in[0];
  const float* Wq = (const float*)d_in[1];
  const float* Wk = (const float*)d_in[2];
  const float* Wv = (const float*)d_in[3];
  const float* Wo = (const float*)d_in[4];
  float* out = (float*)d_out;

  char* wp = (char*)d_ws;
  auto alloc = [&](size_t bytes) -> void* {
    void* p = wp;
    wp += (bytes + 255) & ~(size_t)255;
    return p;
  };

  unsigned short* xb  = (unsigned short*)alloc((size_t)M * C * 2);
  unsigned short* Wqt = (unsigned short*)alloc((size_t)C * Nq * 2);   // [Nq,C]
  unsigned short* Wkt = (unsigned short*)alloc((size_t)C * Nkv * 2);  // [Nkv,C]
  unsigned short* Wvt = (unsigned short*)alloc((size_t)C * Nkv * 2);  // [Nkv,C]
  unsigned short* Wot = (unsigned short*)alloc((size_t)Nq * C * 2);   // [C,Nq]
  float* qf = (float*)alloc((size_t)M * Nq * 4);
  float* kf = (float*)alloc((size_t)M * Nkv * 4);
  float* vf = (float*)alloc((size_t)M * Nkv * 4);
  unsigned short* qbuf = (unsigned short*)alloc((size_t)M * Nq * 2);
  unsigned short* kbuf = (unsigned short*)alloc((size_t)M * Nkv * 2);
  unsigned short* vtb  = (unsigned short*)alloc((size_t)M * Nkv * 2);
  unsigned short* ctx  = (unsigned short*)alloc((size_t)M * Nq * 2);

  cast_bf16_kernel<<<(M * C + 255) / 256, 256, 0, stream>>>(x, xb, M * C);
  cast_transpose_bf16_kernel<<<(C * Nq + 255) / 256, 256, 0, stream>>>(
      Wq, Wqt, C, Nq);
  cast_transpose_bf16_kernel<<<(C * Nkv + 255) / 256, 256, 0, stream>>>(
      Wk, Wkt, C, Nkv);
  cast_transpose_bf16_kernel<<<(C * Nkv + 255) / 256, 256, 0, stream>>>(
      Wv, Wvt, C, Nkv);
  cast_transpose_bf16_kernel<<<(Nq * C + 255) / 256, 256, 0, stream>>>(
      Wo, Wot, Nq, C);

  // Projections
  gemm_bf16_kernel<<<dim3(Nq / 128, M / 64), 128, 0, stream>>>(xb, Wqt, qf, M,
                                                               Nq, C);
  gemm_bf16_kernel<<<dim3(Nkv / 128, M / 64), 128, 0, stream>>>(xb, Wkt, kf, M,
                                                                Nkv, C);
  gemm_bf16_kernel<<<dim3(Nkv / 128, M / 64), 128, 0, stream>>>(xb, Wvt, vf, M,
                                                                Nkv, C);

  // RoPE + layout transforms
  {
    const size_t nq = (size_t)B * T * H * D;
    rope_transpose_kernel<<<(unsigned)((nq + 255) / 256), 256, 0, stream>>>(
        qf, qbuf, B, T, H, D);
    const size_t nk = (size_t)B * T * KV * D;
    rope_transpose_kernel<<<(unsigned)((nk + 255) / 256), 256, 0, stream>>>(
        kf, kbuf, B, T, KV, D);
    v_transpose_kernel<<<(unsigned)((nk + 255) / 256), 256, 0, stream>>>(
        vf, vtb, B, T, KV, D);
  }

  // Flash attention: one wave per (b, h, 16-row q block)
  attn_kernel<<<B * H * (T / 16), 32, 0, stream>>>(qbuf, kbuf, vtb, ctx, T);

  // Output projection
  gemm_bf16_kernel<<<dim3(C / 128, M / 64), 128, 0, stream>>>(ctx, Wot, out, M,
                                                              C, Nq);
}
